// AttnBlock_39127152067099
// MI455X (gfx1250) — compile-verified, matrix operand order checked by reading, not run on hardware
//
#include <hip/hip_runtime.h>

// ---------------------------------------------------------------------------
// AttnBlock for MI455X (gfx1250): GroupNorm + QKV proj + flash attention +
// output proj + residual.  GEMMs via v_wmma_f32_16x16x32_f16; attention K/V
// staged block-wide through LDS with async global->LDS loads (ASYNCcnt).
// ---------------------------------------------------------------------------

typedef _Float16 half8  __attribute__((ext_vector_type(8)));
typedef _Float16 half16 __attribute__((ext_vector_type(16)));
typedef float    float8 __attribute__((ext_vector_type(8)));
typedef int      i32x4  __attribute__((ext_vector_type(4)));

typedef __attribute__((address_space(1))) i32x4 gas_i32x4;   // global int4
typedef __attribute__((address_space(3))) i32x4 las_i32x4;   // LDS int4

#define C_DIM  512
#define HEADS  4
#define HC     128
#define NPIX   4096     // 64*64
#define BATCH  2
#define GROUPS 32
#define CPG    16       // channels per group
#define GN_EPS 1e-6f

#define KPAD   136      // 128 + 8 halves: 272B row stride (68 dwords, 16B aligned)
#define VPAD   40       // 32 + 8 halves:  80B row stride (20 dwords, 16B aligned)

#if defined(__has_builtin)
#if __has_builtin(__builtin_amdgcn_global_load_async_to_lds_b128)
#define HAVE_ASYNC_LDS 1
#endif
#endif

// ---------------------------------------------------------------------------
// Fragment loader: matrix stored row-major [idx][k] (halves), 16B-aligned rows.
// Caller passes p = base + idx*ld + kbase + (lane>=16 ? 8 : 0).
// Elements 0..7  = K kbase+off .. +7   (lo b128)
// Elements 8..15 = K kbase+off+16..+23 (hi b128)
// Matches CDNA5 16-bit A (and mirrored B) 16x32 VGPR layout.
// ---------------------------------------------------------------------------
__device__ __forceinline__ half16 load_frag(const _Float16* p) {
    half8 lo = *(const half8*)(p);
    half8 hi = *(const half8*)(p + 16);
    return __builtin_shufflevector(lo, hi, 0,1,2,3,4,5,6,7,8,9,10,11,12,13,14,15);
}

// Per-lane 16B global->LDS copy (async DMA if available, else via VGPRs).
__device__ __forceinline__ void cp16(const _Float16* g, _Float16* l) {
#ifdef HAVE_ASYNC_LDS
    __builtin_amdgcn_global_load_async_to_lds_b128(
        (gas_i32x4*)(void*)g, (las_i32x4*)(void*)l, 0, 0);
#else
    *(half8*)l = *(const half8*)g;
#endif
}

__device__ __forceinline__ void wait_async4() {
#ifdef HAVE_ASYNC_LDS
#if __has_builtin(__builtin_amdgcn_s_wait_asynccnt)
    __builtin_amdgcn_s_wait_asynccnt(4);
#else
    __asm__ volatile("s_wait_asynccnt 0x4" ::: "memory");
#endif
#endif
}

__device__ __forceinline__ void wait_async0() {
#ifdef HAVE_ASYNC_LDS
#if __has_builtin(__builtin_amdgcn_s_wait_asynccnt)
    __builtin_amdgcn_s_wait_asynccnt(0);
#else
    __asm__ volatile("s_wait_asynccnt 0x0" ::: "memory");
#endif
#endif
}

// ---------------------------------------------------------------------------
// Kernel 1: GroupNorm over [b, c, h, w]; writes fp16 activations transposed to
// [b, n, c] (row-major in c) so GEMM A-fragments are contiguous b128 loads.
// ---------------------------------------------------------------------------
__global__ __launch_bounds__(256) void gn_kernel(
    const float* __restrict__ x, const float* __restrict__ gsc,
    const float* __restrict__ gbi, _Float16* __restrict__ hN) {
    const int bg = blockIdx.x;              // B*GROUPS = 64
    const int b  = bg / GROUPS, g = bg % GROUPS;
    const int c0 = g * CPG;
    const float* xb = x + ((size_t)b * C_DIM + c0) * NPIX;

    __shared__ float red[256], red2[256];
    float s = 0.f, ss = 0.f;
    for (int i = threadIdx.x; i < CPG * NPIX; i += 256) {
        float v = xb[i];
        s += v; ss += v * v;
    }
    red[threadIdx.x] = s; red2[threadIdx.x] = ss;
    __syncthreads();
    for (int off = 128; off > 0; off >>= 1) {
        if ((int)threadIdx.x < off) {
            red[threadIdx.x]  += red[threadIdx.x + off];
            red2[threadIdx.x] += red2[threadIdx.x + off];
        }
        __syncthreads();
    }
    const float invN = 1.0f / (float)(CPG * NPIX);
    const float mean = red[0] * invN;
    const float var  = red2[0] * invN - mean * mean;
    const float inv  = rsqrtf(var + GN_EPS);

    for (int i = threadIdx.x; i < CPG * NPIX; i += 256) {
        int cc = i / NPIX, n = i % NPIX;
        int c  = c0 + cc;
        float v = (xb[i] - mean) * inv * gsc[c] + gbi[c];
        hN[((size_t)b * NPIX + n) * C_DIM + c] = (_Float16)v;
    }
}

// ---------------------------------------------------------------------------
// Kernel 2: fp32 -> fp16 weight conversion. wq/wk/wv concatenated [1536,512],
// wo as [512,512]; both stay [o,c] row-major == WMMA B-fragment layout for
// D = X * W^T (B mirrors A with N in place of M).
// ---------------------------------------------------------------------------
__global__ __launch_bounds__(256) void wcvt_kernel(
    const float* __restrict__ wq, const float* __restrict__ wk,
    const float* __restrict__ wv, const float* __restrict__ wo,
    _Float16* __restrict__ whqkv, _Float16* __restrict__ who) {
    const int i = blockIdx.x * 256 + threadIdx.x;
    const int nqkv = 3 * C_DIM * C_DIM;
    if (i < nqkv) {
        int which = i / (C_DIM * C_DIM);
        int r     = i % (C_DIM * C_DIM);
        const float* w = (which == 0) ? wq : (which == 1) ? wk : wv;
        whqkv[i] = (_Float16)w[r];
    }
    if (i < C_DIM * C_DIM) who[i] = (_Float16)wo[i];
}

// ---------------------------------------------------------------------------
// Kernel 3: fused QKV projection GEMM.  One wave per 16x16 output tile.
// A = hN [N x 512] fp16, B = whqkv rows (o in 0..1535), K-loop of 16 WMMAs.
// Q,K stored [b,h,n,hc]; V stored [b,h,hc,n] for attention B-fragments.
// ---------------------------------------------------------------------------
__global__ __launch_bounds__(256) void qkv_gemm_kernel(
    const _Float16* __restrict__ hN, const _Float16* __restrict__ whqkv,
    const float* __restrict__ bq, const float* __restrict__ bk,
    const float* __restrict__ bv,
    _Float16* __restrict__ qb, _Float16* __restrict__ kb,
    _Float16* __restrict__ vb) {
    const int lane = threadIdx.x & 31;
    const int wave = threadIdx.x >> 5;
    const int tile = blockIdx.x * 8 + wave;         // 2*256*96 = 49152 tiles
    const int ocTiles = (3 * C_DIM) / 16;           // 96
    const int ot = tile % ocTiles;
    const int t2 = tile / ocTiles;
    const int nt = t2 % (NPIX / 16);
    const int b  = t2 / (NPIX / 16);
    const int li = lane & 15, hi = lane >> 4;
    const int khalf = hi * 8;

    const _Float16* arow = hN    + ((size_t)b * NPIX + nt * 16 + li) * C_DIM;
    const _Float16* brow = whqkv + (size_t)(ot * 16 + li) * C_DIM;

    float8 acc = {};
    #pragma unroll
    for (int k0 = 0; k0 < C_DIM; k0 += 32) {
        half16 a  = load_frag(arow + k0 + khalf);
        half16 bf = load_frag(brow + k0 + khalf);
        acc = __builtin_amdgcn_wmma_f32_16x16x32_f16(false, a, false, bf,
                                                     (short)0, acc, false, false);
    }

    const int o    = ot * 16 + li;                  // 0..1535, my D-column
    const int kind = o / C_DIM;                     // 0=q 1=k 2=v
    const int oc   = o % C_DIM;
    const float bias = (kind == 0 ? bq : kind == 1 ? bk : bv)[oc];
    const int head = oc / HC, hc = oc % HC;

    if (kind < 2) {                                 // Q/K: [b,h,n,hc]
        _Float16* dst = (kind == 0 ? qb : kb)
            + (((size_t)b * HEADS + head) * NPIX + nt * 16) * HC + hc;
        #pragma unroll
        for (int v = 0; v < 8; v++) {
            int row = v + 8 * hi;
            dst[(size_t)row * HC] = (_Float16)(acc[v] + bias);
        }
    } else {                                        // V: [b,h,hc,n]
        _Float16* dst = vb
            + (((size_t)b * HEADS + head) * HC + hc) * NPIX + nt * 16;
        #pragma unroll
        for (int v = 0; v < 8; v++) {
            int row = v + 8 * hi;
            dst[row] = (_Float16)(acc[v] + bias);
        }
    }
}

// ---------------------------------------------------------------------------
// Kernel 4: flash attention.  8 waves/block share one (b,h); K/V chunks are
// staged block-wide into double-buffered LDS via async global->LDS DMA, so
// each block streams K/V once instead of 8x.  Per chunk of 32 keys: 8 score
// WMMAs, online softmax (shfl_xor row reductions), D->A transpose of probs
// through padded LDS, 8 PV WMMAs.  Output fp16 to attnO [b, n, c].
// ---------------------------------------------------------------------------
__global__ __launch_bounds__(256) void attn_kernel(
    const _Float16* __restrict__ qb, const _Float16* __restrict__ kb,
    const _Float16* __restrict__ vb, _Float16* __restrict__ attnO) {
    const int lane = threadIdx.x & 31;
    const int wave = threadIdx.x >> 5;
    const int tile = blockIdx.x * 8 + wave;         // B*HEADS*256 = 2048
    const int qt = tile % (NPIX / 16);
    const int bh = tile / (NPIX / 16);              // b*HEADS + h (same for whole block)
    const int li = lane & 15, hi = lane >> 4;
    const int khalf = hi * 8;

    // Double-buffered K (32 x 128) and V^T (128 x 32) chunk tiles + P staging.
    __shared__ __attribute__((aligned(16))) _Float16 kst[2][32][KPAD];
    __shared__ __attribute__((aligned(16))) _Float16 vst[2][HC][VPAD];
    __shared__ __attribute__((aligned(16))) _Float16 pst_all[8][16][40];
    _Float16 (*pst)[40] = pst_all[wave];

    const _Float16* kbase = kb + (size_t)bh * NPIX * HC;   // [n][hc]
    const _Float16* vbase = vb + (size_t)bh * HC * NPIX;   // [hc][n]

    // ---- stage one 32-key chunk into LDS buffer `buf` (4 async ops/wave) ----
    auto stage = [&](int m0, int buf) {
        // K: 32 rows x 256B; 2 rows per instruction; wave owns rows wave*4..+3
        #pragma unroll
        for (int j = 0; j < 2; j++) {
            int row = wave * 4 + j * 2 + (lane >> 4);      // 0..31
            int off = (lane & 15) * 8;                     // halves
            cp16(kbase + (size_t)(m0 + row) * HC + off, &kst[buf][row][off]);
        }
        // V: 128 rows x 64B; 8 rows per instruction; wave owns rows wave*16..+15
        #pragma unroll
        for (int j = 0; j < 2; j++) {
            int row = wave * 16 + j * 8 + (lane >> 2);     // 0..127
            int off = (lane & 3) * 8;                      // halves
            cp16(vbase + (size_t)row * NPIX + m0 + off, &vst[buf][row][off]);
        }
    };

    // Q fragments resident in registers: 16 x 128 fp16 = 4 A-fragments
    const _Float16* qbase = qb + ((size_t)bh * NPIX + qt * 16 + li) * HC;
    half16 qf[4];
    #pragma unroll
    for (int kk = 0; kk < 4; kk++) qf[kk] = load_frag(qbase + kk * 32 + khalf);

    float8 acc[8];
    #pragma unroll
    for (int t = 0; t < 8; t++) acc[t] = (float8){};
    float mi[8], lsum[8];
    #pragma unroll
    for (int v = 0; v < 8; v++) { mi[v] = -1e30f; lsum[v] = 0.f; }

    const float scale = 0.044194173824159216f;      // 512^-0.5 (repo quirk: full C)
    const int NCHUNK = NPIX / 32;                   // 128

    stage(0, 0);                                    // prologue: chunk 0 -> buf 0

    for (int it = 0; it < NCHUNK; it++) {
        const int m0  = it * 32;
        const int cur = it & 1;
        if (it + 1 < NCHUNK) {                      // overlap next chunk's DMA
            stage(m0 + 32, cur ^ 1);
            wait_async4();                          // chunk `it` complete (<=4 pending)
        } else {
            wait_async0();
        }
        __syncthreads();                            // whole tile visible block-wide

        // ---- scores: two adjacent 16x16 key tiles, K-reduced over hc=128 ----
        float8 s0 = {}, s1 = {};
        #pragma unroll
        for (int kk = 0; kk < 4; kk++) {
            half16 b0 = load_frag(&kst[cur][li][kk * 32 + khalf]);
            s0 = __builtin_amdgcn_wmma_f32_16x16x32_f16(false, qf[kk], false, b0,
                                                        (short)0, s0, false, false);
        }
        #pragma unroll
        for (int kk = 0; kk < 4; kk++) {
            half16 b1 = load_frag(&kst[cur][16 + li][kk * 32 + khalf]);
            s1 = __builtin_amdgcn_wmma_f32_16x16x32_f16(false, qf[kk], false, b1,
                                                        (short)0, s1, false, false);
        }

        // ---- online softmax; row v lives in VGPR v across one 16-lane half ----
        #pragma unroll
        for (int v = 0; v < 8; v++) {
            float a0 = s0[v] * scale, a1 = s1[v] * scale;
            float mx = fmaxf(a0, a1);
            #pragma unroll
            for (int m = 8; m >= 1; m >>= 1) mx = fmaxf(mx, __shfl_xor(mx, m, 32));
            float mnew  = fmaxf(mi[v], mx);
            float alpha = __expf(mi[v] - mnew);
            float p0 = __expf(a0 - mnew), p1 = __expf(a1 - mnew);
            float ps = p0 + p1;
            #pragma unroll
            for (int m = 8; m >= 1; m >>= 1) ps += __shfl_xor(ps, m, 32);
            lsum[v] = lsum[v] * alpha + ps;
            mi[v]   = mnew;
            #pragma unroll
            for (int t = 0; t < 8; t++) acc[t][v] *= alpha;
            int row = v + 8 * hi;                   // D layout: my rows
            pst[row][li]      = (_Float16)p0;       // col = key within chunk
            pst[row][16 + li] = (_Float16)p1;
        }

        // D-layout -> A-layout transpose through LDS (wave-private region)
        __asm__ volatile("s_wait_dscnt 0x0" ::: "memory");
        half16 pf = load_frag(&pst[li][khalf]);     // A row = li, K = keys 0..31

        // ---- PV accumulation: 8 hc-column tiles of V^T ----
        #pragma unroll
        for (int t = 0; t < 8; t++) {
            half16 bf = load_frag(&vst[cur][t * 16 + li][khalf]);
            acc[t] = __builtin_amdgcn_wmma_f32_16x16x32_f16(false, pf, false, bf,
                                                            (short)0, acc[t], false, false);
        }
        __syncthreads();                            // done reading buf before refill
    }

    // ---- epilogue: normalize by row sums, write fp16 [b, n, c] ----
    const int b = bh / HEADS, h = bh % HEADS;
    #pragma unroll
    for (int t = 0; t < 8; t++) {
        #pragma unroll
        for (int v = 0; v < 8; v++) {
            int row = qt * 16 + v + 8 * hi;
            int c   = h * HC + t * 16 + li;
            attnO[((size_t)b * NPIX + row) * C_DIM + c] =
                (_Float16)(acc[t][v] / lsum[v]);
        }
    }
}

// ---------------------------------------------------------------------------
// Kernel 5: output projection GEMM + bias + residual, fp32 store [b, c, h, w].
// ---------------------------------------------------------------------------
__global__ __launch_bounds__(256) void out_gemm_kernel(
    const _Float16* __restrict__ attnO, const _Float16* __restrict__ who,
    const float* __restrict__ bo, const float* __restrict__ x,
    float* __restrict__ out) {
    const int lane = threadIdx.x & 31;
    const int wave = threadIdx.x >> 5;
    const int tile = blockIdx.x * 8 + wave;         // 2*256*32 = 16384 tiles
    const int ot = tile % (C_DIM / 16);
    const int t2 = tile / (C_DIM / 16);
    const int nt = t2 % (NPIX / 16);
    const int b  = t2 / (NPIX / 16);
    const int li = lane & 15, hi = lane >> 4;
    const int khalf = hi * 8;

    const _Float16* arow = attnO + ((size_t)b * NPIX + nt * 16 + li) * C_DIM;
    const _Float16* brow = who   + (size_t)(ot * 16 + li) * C_DIM;

    float8 acc = {};
    #pragma unroll
    for (int k0 = 0; k0 < C_DIM; k0 += 32) {
        half16 a  = load_frag(arow + k0 + khalf);
        half16 bf = load_frag(brow + k0 + khalf);
        acc = __builtin_amdgcn_wmma_f32_16x16x32_f16(false, a, false, bf,
                                                     (short)0, acc, false, false);
    }

    const int o = ot * 16 + li;
    const float bias = bo[o];
    const size_t base = ((size_t)b * C_DIM + o) * NPIX + nt * 16;
    #pragma unroll
    for (int v = 0; v < 8; v++) {
        int n = v + 8 * hi;
        out[base + n] = acc[v] + bias + x[base + n];
    }
}

// ---------------------------------------------------------------------------
// Host launcher.  Workspace layout (bytes):
//   hN      [B*N*C fp16]              0        .. 8388608
//   whqkv   [1536*512 fp16]           8388608  .. 9961472
//   who     [512*512 fp16]            9961472  .. 10485760
//   qb      [B*H*N*HC fp16]           10485760 .. 18874368
//   kb                                18874368 .. 27262976
//   vb      (as [b,h,hc,n])           27262976 .. 35651584
//   attnO   [B*N*C fp16]              35651584 .. 44040192
// ---------------------------------------------------------------------------
extern "C" void kernel_launch(void* const* d_in, const int* in_sizes, int n_in,
                              void* d_out, int out_size, void* d_ws, size_t ws_size,
                              hipStream_t stream) {
    const float* x   = (const float*)d_in[0];
    const float* gsc = (const float*)d_in[1];
    const float* gbi = (const float*)d_in[2];
    const float* wq  = (const float*)d_in[3];
    const float* bq  = (const float*)d_in[4];
    const float* wk  = (const float*)d_in[5];
    const float* bk  = (const float*)d_in[6];
    const float* wv  = (const float*)d_in[7];
    const float* bv  = (const float*)d_in[8];
    const float* wo  = (const float*)d_in[9];
    const float* bo  = (const float*)d_in[10];
    float* out = (float*)d_out;

    char* ws = (char*)d_ws;
    _Float16* hN    = (_Float16*)(ws + 0);
    _Float16* whqkv = (_Float16*)(ws + 8388608);
    _Float16* who   = (_Float16*)(ws + 9961472);
    _Float16* qb    = (_Float16*)(ws + 10485760);
    _Float16* kb    = (_Float16*)(ws + 18874368);
    _Float16* vb    = (_Float16*)(ws + 27262976);
    _Float16* attnO = (_Float16*)(ws + 35651584);

    wcvt_kernel<<<3072, 256, 0, stream>>>(wq, wk, wv, wo, whqkv, who);
    gn_kernel<<<BATCH * GROUPS, 256, 0, stream>>>(x, gsc, gbi, hN);
    qkv_gemm_kernel<<<6144, 256, 0, stream>>>(hN, whqkv, bq, bk, bv, qb, kb, vb);
    attn_kernel<<<256, 256, 0, stream>>>(qb, kb, vb, attnO);
    out_gemm_kernel<<<2048, 256, 0, stream>>>(attnO, who, bo, x, out);
}
